// CA_79147657330976
// MI455X (gfx1250) — matrix operand order repaired and checked by reading, not compiled, not run on hardware
//
#include <hip/hip_runtime.h>
#include <math.h>

// ---------------- problem constants ----------------
static constexpr int kN   = 4;
static constexpr int kC   = 1024;   // CIN
static constexpr int kCI  = 512;    // CINT
static constexpr int kH   = 96;
static constexpr int kW   = 96;
static constexpr int kHW  = kH * kW;        // 9216
static constexpr int kM   = kN * kHW;       // 36864 pixels
static constexpr int kD   = 4;
static constexpr int kQ   = (2 * kD + 1) * (2 * kD + 1);  // 81

// ---------------- bf16 helpers ----------------
typedef unsigned short bf16_t;

__device__ __forceinline__ bf16_t f2bf(float f) {
    union { float f; unsigned u; } v; v.f = f;
    unsigned r = v.u + 0x7FFFu + ((v.u >> 16) & 1u);   // round-to-nearest-even
    return (bf16_t)(r >> 16);
}
__device__ __forceinline__ float bf2f(bf16_t b) {
    union { unsigned u; float f; } v; v.u = ((unsigned)b) << 16; return v.f;
}

// ---------------- WMMA vector types (CDNA5 wave32) ----------------
typedef __attribute__((ext_vector_type(16))) __bf16 v16bf;
typedef __attribute__((ext_vector_type(8)))  float  v8f;

union FragBF { v16bf v; unsigned u[8]; };

// =====================================================================
// Weight transpose + fp32 -> bf16 convert:  in [O,K] fp32 -> out [K,O] bf16
// =====================================================================
__global__ void wconv_transpose_bf16(const float* __restrict__ in,
                                     bf16_t* __restrict__ out,
                                     int O, int K) {
    int idx = blockIdx.x * 256 + threadIdx.x;
    if (idx >= O * K) return;
    int k = idx / O;
    int o = idx - k * O;
    out[idx] = f2bf(in[o * K + k]);       // out[k*O+o] = w[o][k]
}

// =====================================================================
// GEMM1: A = X (NCHW fp32, K channels strided by HW), B = bf16 [K,O]
// Out = bf16 NHWC [M,O].  Block: 256 thr (8 waves), tile 128(M) x 64(O).
// Each wave: 16(M) x 64(O) -> 4 WMMAs per K-step (A-fragment reused 4x).
// =====================================================================
__global__ __launch_bounds__(256) void gemm_nchw_bf16(
    const float* __restrict__ X,       // [N, K, HW]
    const bf16_t* __restrict__ Bw,     // [K, Ototal]
    const float* __restrict__ bias,    // [Ototal]
    bf16_t* __restrict__ Out,          // [M, Ototal]
    int K, int Ototal)
{
    __shared__ bf16_t sA[128 * 34];    // [m][k] pad 34 (keeps b32 reads aligned)
    __shared__ bf16_t sB[64 * 34];     // [o][k] pad 34

    const int t    = threadIdx.x;
    const int lane = t & 31;
    const int wave = t >> 5;                    // 0..7 -> M sub-tile
    const int rowBase = blockIdx.x * 128;       // never crosses batch: 9216%128==0
    const int o0      = blockIdx.y * 64;
    const int nb      = rowBase / kHW;
    const int hwBase  = rowBase - nb * kHW;
    const float* Xb = X + (size_t)nb * K * kHW + hwBase;

    v8f acc[4] = {{}, {}, {}, {}};

    const int mrow = lane & 15;
    const int kh   = lane >> 4;

    for (int k0 = 0; k0 < K; k0 += 32) {
        // ---- stage A tile 128x32 (fp32 -> bf16), coalesced along m ----
        #pragma unroll
        for (int i = 0; i < 16; ++i) {
            int idx = i * 256 + t;
            int m  = idx & 127;
            int kk = idx >> 7;                  // 0..31
            sA[m * 34 + kk] = f2bf(Xb[(size_t)(k0 + kk) * kHW + m]);
        }
        // ---- stage B tile 32x64, coalesced along o ----
        #pragma unroll
        for (int i = 0; i < 8; ++i) {
            int idx = i * 256 + t;
            int oo = idx & 63;
            int kk = idx >> 6;                  // 0..31
            sB[oo * 34 + kk] = Bw[(size_t)(k0 + kk) * Ototal + o0 + oo];
        }
        // ---- prefetch next K-tile into cache while WMMAs run ----
        if (k0 + 32 < K) {
            int kk = t >> 3;                    // 0..31
            __builtin_prefetch(&Xb[(size_t)(k0 + 32 + kk) * kHW + (t & 7) * 16], 0, 0);
            __builtin_prefetch(&Bw[(size_t)(k0 + 32 + kk) * Ototal + o0 + (t & 7) * 8], 0, 0);
        }
        __syncthreads();

        // ---- fragments per ISA 16-bit layouts ----
        FragBF fa;
        #pragma unroll
        for (int v = 0; v < 8; ++v) {
            int ka = (v < 4) ? (2 * v + 8 * kh) : (16 + 2 * (v - 4) + 8 * kh);
            fa.u[v] = *(const unsigned*)&sA[(wave * 16 + mrow) * 34 + ka];
        }
        #pragma unroll
        for (int s = 0; s < 4; ++s) {
            FragBF fb;
            #pragma unroll
            for (int v = 0; v < 8; ++v) {
                int kb = (v < 4) ? (2 * v + 8 * kh) : (16 + 2 * (v - 4) + 8 * kh);
                fb.u[v] = *(const unsigned*)&sB[(s * 16 + mrow) * 34 + kb];
            }
            acc[s] = __builtin_amdgcn_wmma_f32_16x16x32_bf16(false, fa.v, false, fb.v,
                                                             (short)0, acc[s], false, false);
        }
        __syncthreads();
    }

    // ---- epilogue: C/D layout -> bf16 NHWC ----
    const int ncol = lane & 15;
    #pragma unroll
    for (int s = 0; s < 4; ++s) {
        int oc = o0 + s * 16 + ncol;
        float bv = bias[oc];
        #pragma unroll
        for (int r = 0; r < 8; ++r) {
            int grow = rowBase + wave * 16 + r + 8 * kh;
            Out[(size_t)grow * Ototal + oc] = f2bf(acc[s][r] + bv);
        }
    }
}

// =====================================================================
// Fused correlation + softmax + assembly.
// One block per 16x16 spatial tile; 81 accumulators per thread in VGPRs.
// =====================================================================
static constexpr int kCC = 8;   // channel chunk

__global__ __launch_bounds__(256) void corr_softmax_assemble(
    const bf16_t* __restrict__ Th,   // theta [M, 512] NHWC bf16
    const bf16_t* __restrict__ Ph,   // phi
    const bf16_t* __restrict__ G,    // g
    bf16_t* __restrict__ Y)          // y [M, 512]
{
    __shared__ float sT[16 * 16 * kCC];
    __shared__ float sP[24 * 24 * kCC];   // padded window tile (also reused for g)

    const int tx = threadIdx.x, ty = threadIdx.y;
    const int t  = ty * 16 + tx;
    const int n  = blockIdx.z;
    const int h0 = blockIdx.y * 16, w0 = blockIdx.x * 16;
    const int h  = h0 + ty, w = w0 + tx;
    const size_t m = (size_t)n * kHW + (size_t)h * kW + w;

    float acc[kQ];
    #pragma unroll
    for (int q = 0; q < kQ; ++q) acc[q] = 0.f;

    // ---------------- phase 1: correlation ----------------
    for (int c0 = 0; c0 < kCI; c0 += kCC) {
        #pragma unroll
        for (int cc = 0; cc < kCC; ++cc)
            sT[t * kCC + cc] = bf2f(Th[m * kCI + c0 + cc]);

        for (int e = t; e < 576; e += 256) {
            int py = e / 24, px = e - py * 24;
            int gh = h0 + py - kD, gw = w0 + px - kD;
            bool in = (gh >= 0) && (gh < kH) && (gw >= 0) && (gw < kW);
            size_t pm = (size_t)n * kHW + (size_t)gh * kW + gw;
            #pragma unroll
            for (int cc = 0; cc < kCC; ++cc)
                sP[e * kCC + cc] = in ? bf2f(Ph[pm * kCI + c0 + cc]) : 0.f;
        }
        __syncthreads();

        const float* tp = &sT[t * kCC];
        #pragma unroll
        for (int q = 0; q < kQ; ++q) {
            int dy = q / 9, dx = q - dy * 9;
            const float* pp = &sP[((ty + dy) * 24 + tx + dx) * kCC];
            float s = 0.f;
            #pragma unroll
            for (int cc = 0; cc < kCC; ++cc) s += tp[cc] * pp[cc];
            acc[q] += s;
        }
        __syncthreads();
    }

    // ---------------- softmax over 81 displacements ----------------
    const float scale = 256.0f / (512.0f * sqrtf(96.0f));  // inv_c * 256/sqrt(W)
    float mx = -3.4e38f;
    #pragma unroll
    for (int q = 0; q < kQ; ++q) { acc[q] *= scale; mx = fmaxf(mx, acc[q]); }
    float sum = 0.f;
    #pragma unroll
    for (int q = 0; q < kQ; ++q) { acc[q] = __expf(acc[q] - mx); sum += acc[q]; }
    float inv = 1.0f / sum;
    #pragma unroll
    for (int q = 0; q < kQ; ++q) acc[q] *= inv;

    // ---------------- phase 2: assembly ----------------
    for (int c0 = 0; c0 < kCI; c0 += kCC) {
        for (int e = t; e < 576; e += 256) {
            int py = e / 24, px = e - py * 24;
            int gh = h0 + py - kD, gw = w0 + px - kD;
            bool in = (gh >= 0) && (gh < kH) && (gw >= 0) && (gw < kW);
            size_t pm = (size_t)n * kHW + (size_t)gh * kW + gw;
            #pragma unroll
            for (int cc = 0; cc < kCC; ++cc)
                sP[e * kCC + cc] = in ? bf2f(G[pm * kCI + c0 + cc]) : 0.f;
        }
        __syncthreads();

        #pragma unroll
        for (int cc = 0; cc < kCC; ++cc) {
            float yv = 0.f;
            #pragma unroll
            for (int q = 0; q < kQ; ++q) {
                int dy = q / 9, dx = q - dy * 9;
                yv += acc[q] * sP[((ty + dy) * 24 + tx + dx) * kCC + cc];
            }
            Y[m * kCI + c0 + cc] = f2bf(yv);
        }
        __syncthreads();
    }
}

// =====================================================================
// GEMM2: A = y (bf16 NHWC [M,512]), B = w_out^T bf16 [512,1024].
// A tile staged with GLOBAL_LOAD_ASYNC_TO_LDS_B128 (ASYNCcnt path):
// y rows are K-contiguous bf16, so the 128x32 tile is 512 aligned 16B
// chunks -> one async b128 per lane x2 iterations, zero VGPR traffic.
// Tile 128(M) x 64(O); epilogue: + b_out + residual x -> fp32 NCHW.
// =====================================================================
__global__ __launch_bounds__(256) void gemm_out_residual(
    const bf16_t* __restrict__ Y,      // [M, K]
    const bf16_t* __restrict__ Bw,     // [K, Ototal]
    const float* __restrict__ bias,    // [Ototal]
    const float* __restrict__ Xres,    // [N, Ototal, HW]
    float* __restrict__ Outp,          // [N, Ototal, HW]
    int K, int Ototal)
{
    __shared__ bf16_t sA[128 * 40];    // [m][k] pad 40 -> 80B rows, 16B-aligned chunks
    __shared__ bf16_t sB[64 * 34];     // [o][k] pad 34

    const int t    = threadIdx.x;
    const int lane = t & 31;
    const int wave = t >> 5;
    const int rowBase = blockIdx.x * 128;
    const int o0      = blockIdx.y * 64;

    v8f acc[4] = {{}, {}, {}, {}};
    const int mrow = lane & 15;
    const int kh   = lane >> 4;

    for (int k0 = 0; k0 < K; k0 += 32) {
        // ---- A tile 128x32 bf16 via async copy to LDS (2 x b128/lane) ----
        #pragma unroll
        for (int i = 0; i < 2; ++i) {
            int idx = i * 256 + t;             // 0..511 chunks of 16B
            int m   = idx >> 2;                // 0..127
            int c   = idx & 3;                 // 16B chunk within the 64B row
            const bf16_t* gp = &Y[(size_t)(rowBase + m) * K + k0 + c * 8];
            unsigned lp = (unsigned)(size_t)&sA[m * 40 + c * 8];   // LDS byte offset (low 32b)
            asm volatile("global_load_async_to_lds_b128 %0, %1, off"
                         :: "v"(lp), "v"(gp) : "memory");
        }
        // ---- stage B tile 32x64 (transposed layout), coalesced along o ----
        #pragma unroll
        for (int i = 0; i < 8; ++i) {
            int idx = i * 256 + t;
            int oo = idx & 63;
            int kk = idx >> 6;
            sB[oo * 34 + kk] = Bw[(size_t)(k0 + kk) * Ototal + o0 + oo];
        }
        if (k0 + 32 < K) {
            __builtin_prefetch(&Bw[(size_t)(k0 + 32 + (t >> 3)) * Ototal + o0 + (t & 7) * 8], 0, 0);
        }
        asm volatile("s_wait_asynccnt 0x0" ::: "memory");
        __syncthreads();

        FragBF fa;
        #pragma unroll
        for (int v = 0; v < 8; ++v) {
            int ka = (v < 4) ? (2 * v + 8 * kh) : (16 + 2 * (v - 4) + 8 * kh);
            fa.u[v] = *(const unsigned*)&sA[(wave * 16 + mrow) * 40 + ka];
        }
        #pragma unroll
        for (int s = 0; s < 4; ++s) {
            FragBF fb;
            #pragma unroll
            for (int v = 0; v < 8; ++v) {
                int kb = (v < 4) ? (2 * v + 8 * kh) : (16 + 2 * (v - 4) + 8 * kh);
                fb.u[v] = *(const unsigned*)&sB[(s * 16 + mrow) * 34 + kb];
            }
            acc[s] = __builtin_amdgcn_wmma_f32_16x16x32_bf16(false, fa.v, false, fb.v,
                                                             (short)0, acc[s], false, false);
        }
        __syncthreads();
    }

    const int ncol = lane & 15;
    #pragma unroll
    for (int s = 0; s < 4; ++s) {
        int oc = o0 + s * 16 + ncol;
        float bv = bias[oc];
        #pragma unroll
        for (int r = 0; r < 8; ++r) {
            int grow = rowBase + wave * 16 + r + 8 * kh;   // pixel index
            int nb   = grow / kHW;
            int hw   = grow - nb * kHW;
            size_t i0 = (size_t)nb * Ototal * kHW + (size_t)oc * kHW + hw;
            Outp[i0] = acc[s][r] + bv + Xres[i0];
        }
    }
}

// =====================================================================
// launch
// =====================================================================
extern "C" void kernel_launch(void* const* d_in, const int* in_sizes, int n_in,
                              void* d_out, int out_size, void* d_ws, size_t ws_size,
                              hipStream_t stream) {
    const float* x       = (const float*)d_in[0];
    const float* x_ref   = (const float*)d_in[1];
    const float* w_g     = (const float*)d_in[2];
    const float* b_g     = (const float*)d_in[3];
    const float* w_theta = (const float*)d_in[4];
    const float* b_theta = (const float*)d_in[5];
    const float* w_phi   = (const float*)d_in[6];
    const float* b_phi   = (const float*)d_in[7];
    const float* w_out   = (const float*)d_in[8];
    const float* b_out   = (const float*)d_in[9];
    float* outp = (float*)d_out;

    // workspace carve-out (bf16 everywhere), ~155 MB total
    char* ws = (char*)d_ws;
    size_t off = 0;
    const size_t wKOsz = (size_t)kC * kCI * sizeof(bf16_t);       // 1 MB each
    bf16_t* wtT = (bf16_t*)(ws + off); off += wKOsz;              // w_theta^T [1024,512]
    bf16_t* wpT = (bf16_t*)(ws + off); off += wKOsz;              // w_phi^T
    bf16_t* wgT = (bf16_t*)(ws + off); off += wKOsz;              // w_g^T
    bf16_t* woT = (bf16_t*)(ws + off); off += wKOsz;              // w_out^T [512,1024]
    const size_t featSz = (size_t)kM * kCI * sizeof(bf16_t);      // ~37.7 MB each
    bf16_t* thB = (bf16_t*)(ws + off); off += featSz;             // theta
    bf16_t* phB = (bf16_t*)(ws + off); off += featSz;             // phi
    bf16_t* gB  = (bf16_t*)(ws + off); off += featSz;             // g
    bf16_t* yB  = (bf16_t*)(ws + off); off += featSz;             // y

    // 1) weight transposes -> bf16 [K,O]
    {
        int total = kC * kCI;
        int blocks = (total + 255) / 256;
        wconv_transpose_bf16<<<blocks, 256, 0, stream>>>(w_theta, wtT, kCI, kC);
        wconv_transpose_bf16<<<blocks, 256, 0, stream>>>(w_phi,   wpT, kCI, kC);
        wconv_transpose_bf16<<<blocks, 256, 0, stream>>>(w_g,     wgT, kCI, kC);
        wconv_transpose_bf16<<<blocks, 256, 0, stream>>>(w_out,   woT, kC, kCI);
    }

    // 2) theta / phi / g GEMMs (WMMA bf16), output NHWC bf16
    {
        dim3 grid(kM / 128, kCI / 64);   // 288 x 8
        gemm_nchw_bf16<<<grid, 256, 0, stream>>>(x,     wtT, b_theta, thB, kC, kCI);
        gemm_nchw_bf16<<<grid, 256, 0, stream>>>(x_ref, wpT, b_phi,   phB, kC, kCI);
        gemm_nchw_bf16<<<grid, 256, 0, stream>>>(x_ref, wgT, b_g,     gB,  kC, kCI);
    }

    // 3) fused correlation + softmax + assembly
    {
        dim3 grid(kW / 16, kH / 16, kN); // 6 x 6 x 4
        dim3 block(16, 16);
        corr_softmax_assemble<<<grid, block, 0, stream>>>(thB, phB, gB, yB);
    }

    // 4) output GEMM + bias + residual -> fp32 NCHW
    {
        dim3 grid(kM / 128, kC / 64);    // 288 x 16
        gemm_out_residual<<<grid, 256, 0, stream>>>(yB, woT, b_out, x, outp, kCI, kC);
    }
}
